// Seq2SeqTrain_69810398429896
// MI455X (gfx1250) — matrix operand order repaired and checked by reading, not compiled, not run on hardware
//
#include <hip/hip_runtime.h>
#include <hip/hip_bf16.h>

// ---------------------------------------------------------------------------
// Seq2Seq (biGRU encoder + attention GRU decoder) forward for MI455X/gfx1250.
// All GEMMs run through v_wmma_f32_16x16x32_bf16 (fp32 accumulate).
// Weights converted fp32 -> bf16 once per call; W_out (114.7MB bf16) becomes
// L2-resident (192MB L2) so the 31 decoder-step output projections hit L2
// instead of streaming 229MB fp32 from HBM each step.
// ---------------------------------------------------------------------------

typedef __bf16 bf16;
typedef __attribute__((ext_vector_type(16))) __bf16 v16bf;
typedef __attribute__((ext_vector_type(8)))  float  v8f;

#define S_LEN 64
#define BATCH 128
#define T_LEN 32
#define E_DIM 256
#define HE_DIM 512
#define HD_DIM 512
#define VOCAB 32000

// ------------------------- generic WMMA GEMM -------------------------------
// C[M,N] = act( A[M,K](bf16,row-major,lda) @ W[N,K](bf16,row-major,ldw)^T + bias )
// One wave -> 16(M) x 64(N) tile: A fragment reused across 4 accumulators.
// Requires M%16==0, N%64==0, K%32==0 (true for every GEMM in this model).
__global__ void gemm_wmma_bf16(const bf16* __restrict__ A, int lda,
                               const bf16* __restrict__ W, int ldw,
                               const float* __restrict__ bias,
                               float* __restrict__ outF, int ldof,
                               bf16* __restrict__ outH, int ldoh,
                               int M, int N, int K, int act)
{
    const int lane  = threadIdx.x & 31;
    const int wave  = (int)((blockIdx.x * blockDim.x + threadIdx.x) >> 5);
    const int tilesN = N >> 6;
    const int tm = (wave / tilesN) << 4;
    const int tn = (wave % tilesN) << 6;
    if (tm >= M) return;

    const int hi = lane >> 4;   // half-wave select
    const int lr = lane & 15;

    v8f acc0 = {}, acc1 = {}, acc2 = {}, acc3 = {};

    // A: lane holds row m=tm+lr; elements e -> k = hi*16 + e (contiguous)
    const bf16* arow  = A + (size_t)(tm + lr) * lda + hi * 16;
    // B: lane holds W row n (=B column); elements e -> k = hi*16 + e (contiguous)
    const bf16* wrow0 = W + (size_t)(tn + lr) * ldw + hi * 16;
    const bf16* wrow1 = wrow0 + (size_t)16 * ldw;
    const bf16* wrow2 = wrow0 + (size_t)32 * ldw;
    const bf16* wrow3 = wrow0 + (size_t)48 * ldw;

    for (int k0 = 0; k0 < K; k0 += 32) {
        v16bf af, b0, b1, b2, b3;
#pragma unroll
        for (int e = 0; e < 8; ++e) { af[e] = arow[k0 + e]; af[e + 8] = arow[k0 + 8 + e]; }
#pragma unroll
        for (int e = 0; e < 16; ++e) {
            b0[e] = wrow0[k0 + e]; b1[e] = wrow1[k0 + e];
            b2[e] = wrow2[k0 + e]; b3[e] = wrow3[k0 + e];
        }
        acc0 = __builtin_amdgcn_wmma_f32_16x16x32_bf16(false, af, false, b0, (short)0, acc0, false, false);
        acc1 = __builtin_amdgcn_wmma_f32_16x16x32_bf16(false, af, false, b1, (short)0, acc1, false, false);
        acc2 = __builtin_amdgcn_wmma_f32_16x16x32_bf16(false, af, false, b2, (short)0, acc2, false, false);
        acc3 = __builtin_amdgcn_wmma_f32_16x16x32_bf16(false, af, false, b3, (short)0, acc3, false, false);
    }

    // C/D layout: element r -> M = tm + 8*hi + r, N = tn + q*16 + lr
    auto store_tile = [&](v8f acc, int q) {
        int n = tn + q * 16 + lr;
        float bv = bias ? bias[n] : 0.0f;
#pragma unroll
        for (int r = 0; r < 8; ++r) {
            int m = tm + (hi << 3) + r;
            float v = acc[r] + bv;
            if (act) v = tanhf(v);
            if (outF) outF[(size_t)m * ldof + n] = v;
            if (outH) outH[(size_t)m * ldoh + n] = (bf16)v;
        }
    };
    store_tile(acc0, 0); store_tile(acc1, 1); store_tile(acc2, 2); store_tile(acc3, 3);
}

// ------------------------- pointwise / helper kernels ----------------------
__global__ void cvt_bf16_kernel(const float* __restrict__ s, bf16* __restrict__ d, long n)
{
    long i = (long)blockIdx.x * blockDim.x + threadIdx.x;
    long stride = (long)gridDim.x * blockDim.x;
    for (; i < n; i += stride) d[i] = (bf16)s[i];
}

// dst[r*cols+c] = bf16(src[r*srcStride + srcOff + c])   (used to split W_attn)
__global__ void cvt_bf16_strided(const float* __restrict__ s, bf16* __restrict__ d,
                                 int rows, int cols, int srcStride, int srcOff)
{
    long i = (long)blockIdx.x * blockDim.x + threadIdx.x;
    long n = (long)rows * cols;
    long stride = (long)gridDim.x * blockDim.x;
    for (; i < n; i += stride) {
        int r = (int)(i / cols), c = (int)(i % cols);
        d[i] = (bf16)s[(size_t)r * srcStride + srcOff + c];
    }
}

__global__ void gather_src_kernel(const int* __restrict__ tok, const float* __restrict__ emb,
                                  bf16* __restrict__ X)  // X: (S*B, E)
{
    long i = (long)blockIdx.x * blockDim.x + threadIdx.x;
    if (i >= (long)S_LEN * BATCH * E_DIM) return;
    long row = i >> 8; int e = (int)(i & 255);
    X[i] = (bf16)emb[(size_t)tok[row] * E_DIM + e];
}

__global__ void fill_f32_kernel(float* p, float v, long n)
{
    long i = (long)blockIdx.x * blockDim.x + threadIdx.x;
    if (i < n) p[i] = v;
}

__global__ void fill_bf16_kernel(bf16* p, long n)
{
    long i = (long)blockIdx.x * blockDim.x + threadIdx.x;
    if (i < n) p[i] = (bf16)0.0f;
}

// Fused GRU gates: h = (1-z)*n + z*h ; dual bf16 stores for downstream WMMA.
__global__ void gru_fuse_kernel(const float* __restrict__ gi, const float* __restrict__ gh,
                                float* __restrict__ h,
                                bf16* d1, int ld1, bf16* d2, int ld2, int H)
{
    int i = blockIdx.x * blockDim.x + threadIdx.x;
    if (i >= BATCH * H) return;
    int b = i / H, j = i % H;
    const float* gib = gi + (size_t)b * 3 * H;
    const float* ghb = gh + (size_t)b * 3 * H;
    float r  = 1.0f / (1.0f + expf(-(gib[j]       + ghb[j])));
    float z  = 1.0f / (1.0f + expf(-(gib[H + j]   + ghb[H + j])));
    float nn = tanhf(gib[2 * H + j] + r * ghb[2 * H + j]);
    float hv = (1.0f - z) * nn + z * h[i];
    h[i] = hv;
    bf16 hb = (bf16)hv;
    if (d1) d1[(size_t)b * ld1 + j] = hb;
    if (d2) d2[(size_t)b * ld2 + j] = hb;
}

__global__ void concat_h_kernel(const float* __restrict__ hf, const float* __restrict__ hb,
                                bf16* __restrict__ dst)  // (B, 2*HE)
{
    int i = blockIdx.x * blockDim.x + threadIdx.x;
    if (i >= BATCH * 2 * HE_DIM) return;
    int b = i >> 10, c = i & 1023;
    float v = (c < HE_DIM) ? hf[b * HE_DIM + c] : hb[b * HE_DIM + (c - HE_DIM)];
    dst[i] = (bf16)v;
}

// Gather decoder-step embedding into GRU input (x_d[:, :256]) and the output
// projection concat (cat[:, 1536:1792]).
__global__ void dec_emb_kernel(const int* __restrict__ trg_t, const float* __restrict__ emb,
                               bf16* __restrict__ xd, bf16* __restrict__ cat)
{
    int i = blockIdx.x * blockDim.x + threadIdx.x;
    if (i >= BATCH * E_DIM) return;
    int b = i >> 8, e = i & 255;
    bf16 v = (bf16)emb[(size_t)trg_t[b] * E_DIM + e];
    xd[(size_t)b * 1280 + e] = v;
    cat[(size_t)b * 1792 + 1536 + e] = v;
}

// scores[b,s] = mask ? sum_d v[d]*tanh(enc_proj[b,s,d] + hWah[b,d]) : -1e10
__global__ void attn_scores_kernel(const float* __restrict__ enc_proj,
                                   const float* __restrict__ hWah,
                                   const float* __restrict__ v_attn,
                                   const int* __restrict__ src,
                                   float* __restrict__ scores)
{
    int bs = blockIdx.x;              // B*S blocks
    int b = bs >> 6, s = bs & 63;
    const float* ep = enc_proj + ((size_t)b * S_LEN + s) * HD_DIM;
    const float* hw = hWah + (size_t)b * HD_DIM;
    float part = 0.0f;
    for (int d = threadIdx.x; d < HD_DIM; d += blockDim.x)
        part += v_attn[d] * tanhf(ep[d] + hw[d]);
    __shared__ float red[256];
    red[threadIdx.x] = part; __syncthreads();
    for (int off = 128; off > 0; off >>= 1) {
        if ((int)threadIdx.x < off) red[threadIdx.x] += red[threadIdx.x + off];
        __syncthreads();
    }
    if (threadIdx.x == 0) {
        bool m = src[s * BATCH + b] != 0;   // PAD == 0
        scores[b * S_LEN + s] = m ? red[0] : -1e10f;
    }
}

// softmax over s (64) then context w[b,:] = sum_s a[s]*enc_b[b,s,:]; dual bf16 store.
__global__ void attn_softmax_w_kernel(const float* __restrict__ scores,
                                      const bf16* __restrict__ enc_b,
                                      bf16* __restrict__ d1, int ld1,
                                      bf16* __restrict__ d2, int ld2)
{
    int b = blockIdx.x;               // 128 blocks x 128 threads
    int tid = threadIdx.x;
    __shared__ float a[S_LEN];
    __shared__ float red[128];
    float sc = (tid < S_LEN) ? scores[b * S_LEN + tid] : -1e30f;
    red[tid] = sc; __syncthreads();
    for (int off = 64; off > 0; off >>= 1) {
        if (tid < off) red[tid] = fmaxf(red[tid], red[tid + off]);
        __syncthreads();
    }
    float mx = red[0]; __syncthreads();
    float ex = (tid < S_LEN) ? expf(sc - mx) : 0.0f;
    red[tid] = ex; __syncthreads();
    for (int off = 64; off > 0; off >>= 1) {
        if (tid < off) red[tid] += red[tid + off];
        __syncthreads();
    }
    float inv = 1.0f / red[0];
    if (tid < S_LEN) a[tid] = ex * inv;
    __syncthreads();
    const bf16* eb = enc_b + (size_t)b * S_LEN * 1024;
    for (int c = tid; c < 1024; c += 128) {
        float acc = 0.0f;
        for (int s = 0; s < S_LEN; ++s) acc += a[s] * (float)eb[(size_t)s * 1024 + c];
        bf16 hv = (bf16)acc;
        d1[(size_t)b * ld1 + c] = hv;
        d2[(size_t)b * ld2 + c] = hv;
    }
}

__global__ void log_softmax_kernel(const float* __restrict__ logits, float* __restrict__ out)
{
    int b = blockIdx.x;               // 128 blocks x 256 threads
    int tid = threadIdx.x;
    const float* lg = logits + (size_t)b * VOCAB;
    __shared__ float red[256];
    float mx = -1e30f;
    for (int v = tid; v < VOCAB; v += 256) mx = fmaxf(mx, lg[v]);
    red[tid] = mx; __syncthreads();
    for (int off = 128; off > 0; off >>= 1) {
        if (tid < off) red[tid] = fmaxf(red[tid], red[tid + off]);
        __syncthreads();
    }
    mx = red[0]; __syncthreads();
    float sum = 0.0f;
    for (int v = tid; v < VOCAB; v += 256) sum += expf(lg[v] - mx);
    red[tid] = sum; __syncthreads();
    for (int off = 128; off > 0; off >>= 1) {
        if (tid < off) red[tid] += red[tid + off];
        __syncthreads();
    }
    float lse = mx + logf(red[0]);
    for (int v = tid; v < VOCAB; v += 256) out[(size_t)b * VOCAB + v] = lg[v] - lse;
}

// Last output slice: 0 everywhere, 100 at EOS (==2)
__global__ void fill_eos_kernel(float* __restrict__ out)
{
    long i = (long)blockIdx.x * blockDim.x + threadIdx.x;
    if (i >= (long)BATCH * VOCAB) return;
    out[i] = ((i % VOCAB) == 2) ? 100.0f : 0.0f;
}

// ------------------------------- host side ---------------------------------
extern "C" void kernel_launch(void* const* d_in, const int* in_sizes, int n_in,
                              void* d_out, int out_size, void* d_ws, size_t ws_size,
                              hipStream_t stream)
{
    (void)in_sizes; (void)n_in; (void)out_size; (void)ws_size;
    const int*   src     = (const int*)d_in[0];
    /* src_len (d_in[1]) unused: all sequences full length; mask from src!=PAD */
    const int*   trg     = (const int*)d_in[2];
    const float* emb_enc = (const float*)d_in[3];
    const float* emb_dec = (const float*)d_in[4];
    const float* Wih_f   = (const float*)d_in[5];
    const float* Whh_f   = (const float*)d_in[6];
    const float* bih_f   = (const float*)d_in[7];
    const float* bhh_f   = (const float*)d_in[8];
    const float* Wih_b   = (const float*)d_in[9];
    const float* Whh_b   = (const float*)d_in[10];
    const float* bih_b   = (const float*)d_in[11];
    const float* bhh_b   = (const float*)d_in[12];
    const float* W_fc    = (const float*)d_in[13];
    const float* b_fc    = (const float*)d_in[14];
    const float* W_attn  = (const float*)d_in[15];
    const float* b_attn  = (const float*)d_in[16];
    const float* v_attn  = (const float*)d_in[17];
    const float* Wih_d   = (const float*)d_in[18];
    const float* Whh_d   = (const float*)d_in[19];
    const float* bih_d   = (const float*)d_in[20];
    const float* bhh_d   = (const float*)d_in[21];
    const float* W_out   = (const float*)d_in[22];
    const float* b_out   = (const float*)d_in[23];
    float* out = (float*)d_out;

    // Bump allocator over workspace (~185 MB total)
    char* base = (char*)d_ws;
    size_t off = 0;
    auto alloc = [&](size_t bytes) -> void* {
        void* r = base + off;
        off += (bytes + 255) & ~(size_t)255;
        return r;
    };
    bf16* Wih_f_h = (bf16*)alloc((size_t)1536 * 256 * 2);
    bf16* Whh_f_h = (bf16*)alloc((size_t)1536 * 512 * 2);
    bf16* Wih_b_h = (bf16*)alloc((size_t)1536 * 256 * 2);
    bf16* Whh_b_h = (bf16*)alloc((size_t)1536 * 512 * 2);
    bf16* W_fc_h  = (bf16*)alloc((size_t)512 * 1024 * 2);
    bf16* W_a_h_h = (bf16*)alloc((size_t)512 * 512 * 2);
    bf16* W_a_e_h = (bf16*)alloc((size_t)512 * 1024 * 2);
    bf16* Wih_d_h = (bf16*)alloc((size_t)1536 * 1280 * 2);
    bf16* Whh_d_h = (bf16*)alloc((size_t)1536 * 512 * 2);
    bf16* W_out_h = (bf16*)alloc((size_t)VOCAB * 1792 * 2);          // 114.7 MB -> L2 resident
    bf16* X_src   = (bf16*)alloc((size_t)S_LEN * BATCH * E_DIM * 2);
    bf16* enc_b   = (bf16*)alloc((size_t)BATCH * S_LEN * 1024 * 2);  // (B,S,2HE)
    float* enc_proj = (float*)alloc((size_t)BATCH * S_LEN * HD_DIM * 4);
    float* h_f    = (float*)alloc((size_t)BATCH * HE_DIM * 4);
    bf16*  h_f_h  = (bf16*)alloc((size_t)BATCH * HE_DIM * 2);
    float* h_b    = (float*)alloc((size_t)BATCH * HE_DIM * 4);
    bf16*  h_b_h  = (bf16*)alloc((size_t)BATCH * HE_DIM * 2);
    float* gi     = (float*)alloc((size_t)BATCH * 1536 * 4);
    float* gh     = (float*)alloc((size_t)BATCH * 1536 * 4);
    bf16*  cat_h  = (bf16*)alloc((size_t)BATCH * 1024 * 2);
    float* h_dec  = (float*)alloc((size_t)BATCH * HD_DIM * 4);
    bf16*  h_dec_h= (bf16*)alloc((size_t)BATCH * HD_DIM * 2);
    float* hWah   = (float*)alloc((size_t)BATCH * HD_DIM * 4);
    float* scores = (float*)alloc((size_t)BATCH * S_LEN * 4);
    bf16*  x_d    = (bf16*)alloc((size_t)BATCH * 1280 * 2);
    bf16*  catd   = (bf16*)alloc((size_t)BATCH * 1792 * 2);
    float* logits = (float*)alloc((size_t)BATCH * VOCAB * 4);

    auto cvt = [&](const float* s, bf16* d, long n) {
        long b = (n + 255) / 256; if (b > 4096) b = 4096;
        cvt_bf16_kernel<<<dim3((unsigned)b), dim3(256), 0, stream>>>(s, d, n);
    };
    auto gemm = [&](const bf16* A, int lda, const bf16* W, int ldw, const float* bias,
                    float* oF, int ldof, bf16* oH, int ldoh, int M, int N, int K, int act) {
        int waves = (M / 16) * (N / 64);
        int blocks = (waves * 32 + 255) / 256;
        gemm_wmma_bf16<<<dim3(blocks), dim3(256), 0, stream>>>(A, lda, W, ldw, bias,
                                                               oF, ldof, oH, ldoh, M, N, K, act);
    };

    // ---- weight / embedding conversion -----------------------------------
    cvt(Wih_f, Wih_f_h, 1536L * 256);
    cvt(Whh_f, Whh_f_h, 1536L * 512);
    cvt(Wih_b, Wih_b_h, 1536L * 256);
    cvt(Whh_b, Whh_b_h, 1536L * 512);
    cvt(W_fc,  W_fc_h,  512L * 1024);
    cvt_bf16_strided<<<dim3(1024), dim3(256), 0, stream>>>(W_attn, W_a_h_h, 512, 512, 1536, 0);
    cvt_bf16_strided<<<dim3(2048), dim3(256), 0, stream>>>(W_attn, W_a_e_h, 512, 1024, 1536, 512);
    cvt(Wih_d, Wih_d_h, 1536L * 1280);
    cvt(Whh_d, Whh_d_h, 1536L * 512);
    cvt(W_out, W_out_h, (long)VOCAB * 1792);
    gather_src_kernel<<<dim3((S_LEN * BATCH * E_DIM + 255) / 256), dim3(256), 0, stream>>>(
        src, emb_enc, X_src);

    // ---- encoder: bidirectional GRU --------------------------------------
    fill_f32_kernel<<<dim3(256), dim3(256), 0, stream>>>(h_f, 0.0f, (long)BATCH * HE_DIM);
    fill_f32_kernel<<<dim3(256), dim3(256), 0, stream>>>(h_b, 0.0f, (long)BATCH * HE_DIM);
    fill_bf16_kernel<<<dim3(256), dim3(256), 0, stream>>>(h_f_h, (long)BATCH * HE_DIM);
    fill_bf16_kernel<<<dim3(256), dim3(256), 0, stream>>>(h_b_h, (long)BATCH * HE_DIM);

    const int fuseBlocks = (BATCH * HE_DIM + 255) / 256;
    for (int t = 0; t < S_LEN; ++t) {
        // forward direction
        gemm(X_src + (size_t)t * BATCH * E_DIM, E_DIM, Wih_f_h, E_DIM, bih_f,
             gi, 1536, nullptr, 0, BATCH, 1536, E_DIM, 0);
        gemm(h_f_h, HE_DIM, Whh_f_h, HE_DIM, bhh_f,
             gh, 1536, nullptr, 0, BATCH, 1536, HE_DIM, 0);
        gru_fuse_kernel<<<dim3(fuseBlocks), dim3(256), 0, stream>>>(
            gi, gh, h_f, enc_b + (size_t)t * 1024, S_LEN * 1024, h_f_h, HE_DIM, HE_DIM);
        // backward direction (reads time S-1-t)
        int tb = S_LEN - 1 - t;
        gemm(X_src + (size_t)tb * BATCH * E_DIM, E_DIM, Wih_b_h, E_DIM, bih_b,
             gi, 1536, nullptr, 0, BATCH, 1536, E_DIM, 0);
        gemm(h_b_h, HE_DIM, Whh_b_h, HE_DIM, bhh_b,
             gh, 1536, nullptr, 0, BATCH, 1536, HE_DIM, 0);
        gru_fuse_kernel<<<dim3(fuseBlocks), dim3(256), 0, stream>>>(
            gi, gh, h_b, enc_b + (size_t)tb * 1024 + 512, S_LEN * 1024, h_b_h, HE_DIM, HE_DIM);
    }

    // hid = tanh([hf,hb] @ W_fc^T + b_fc)  -> decoder initial state
    concat_h_kernel<<<dim3((BATCH * 1024 + 255) / 256), dim3(256), 0, stream>>>(h_f, h_b, cat_h);
    gemm(cat_h, 1024, W_fc_h, 1024, b_fc, h_dec, HD_DIM, h_dec_h, HD_DIM,
         BATCH, HD_DIM, 1024, /*tanh*/1);

    // enc_proj = enc_b @ W_a_e^T + b_attn   (8192 x 1024 -> 512)
    gemm(enc_b, 1024, W_a_e_h, 1024, b_attn, enc_proj, HD_DIM, nullptr, 0,
         BATCH * S_LEN, HD_DIM, 1024, 0);

    // ---- decoder loop -----------------------------------------------------
    for (int t = 0; t < T_LEN - 1; ++t) {
        dec_emb_kernel<<<dim3((BATCH * E_DIM + 255) / 256), dim3(256), 0, stream>>>(
            trg + (size_t)t * BATCH, emb_dec, x_d, catd);
        // h @ W_a_h^T
        gemm(h_dec_h, HD_DIM, W_a_h_h, HD_DIM, nullptr, hWah, HD_DIM, nullptr, 0,
             BATCH, HD_DIM, HD_DIM, 0);
        attn_scores_kernel<<<dim3(BATCH * S_LEN), dim3(256), 0, stream>>>(
            enc_proj, hWah, v_attn, src, scores);
        attn_softmax_w_kernel<<<dim3(BATCH), dim3(128), 0, stream>>>(
            scores, enc_b, x_d + 256, 1280, catd + 512, 1792);
        // decoder GRU
        gemm(x_d, 1280, Wih_d_h, 1280, bih_d, gi, 1536, nullptr, 0, BATCH, 1536, 1280, 0);
        gemm(h_dec_h, HD_DIM, Whh_d_h, HD_DIM, bhh_d, gh, 1536, nullptr, 0, BATCH, 1536, HD_DIM, 0);
        gru_fuse_kernel<<<dim3(fuseBlocks), dim3(256), 0, stream>>>(
            gi, gh, h_dec, h_dec_h, HD_DIM, catd, 1792, HD_DIM);
        // logits = [h2, w, e] @ W_out^T + b_out   (the 455-GFLOP hot loop)
        gemm(catd, 1792, W_out_h, 1792, b_out, logits, VOCAB, nullptr, 0,
             BATCH, VOCAB, 1792, 0);
        log_softmax_kernel<<<dim3(BATCH), dim3(256), 0, stream>>>(
            logits, out + (size_t)t * BATCH * VOCAB);
    }

    // last timestep: zeros with EOS=100
    fill_eos_kernel<<<dim3((BATCH * VOCAB + 255) / 256), dim3(256), 0, stream>>>(
        out + (size_t)(T_LEN - 1) * BATCH * VOCAB);
}
